// LXMERT_REX_73186242724173
// MI455X (gfx1250) — matrix operand order repaired and checked by reading, not compiled, not run on hardware
//
// LXMERT-REX forward for MI455X (gfx1250): bf16 WMMA GEMMs (f32 accumulate),
// f32 elementwise/softmax/argmax. Compile-only target; wave32 / WMMA path.
#include <hip/hip_runtime.h>
#include <cstdint>
#include <cstddef>

#define KB 256    // batch
#define KR 36     // regions
#define KD 2048   // image dim
#define KH 768    // hidden
#define KV 2001   // vocab
#define KA 2000   // answers
#define KT 12     // steps

typedef __attribute__((ext_vector_type(16))) __bf16 v16bf;
typedef __attribute__((ext_vector_type(8)))  float  v8f;

union HalfVec { uint4 u[2]; v16bf v; };

__device__ __forceinline__ unsigned short f2bf(float f) {
  unsigned int u = __float_as_uint(f);
  u += 0x7fffu + ((u >> 16) & 1u);        // round-to-nearest-even
  return (unsigned short)(u >> 16);
}
__device__ __forceinline__ float sigmoidf_(float x) { return 1.0f / (1.0f + expf(-x)); }

// ---------------------------------------------------------------------------
// GEMM: C[M,N] = act(A_bf16[M,K](lda) * WT_bf16[N,K]^T + bias[N])
// one wave = one 16x16 tile; block = 8 waves = 32x64 macro-tile.
// act: 0=none, 1=tanh, 2=sigmoid
// ---------------------------------------------------------------------------
__global__ __launch_bounds__(256)
void k_gemm_bf16(const unsigned short* __restrict__ A, int lda,
                 const unsigned short* __restrict__ WT,
                 const float* __restrict__ bias,
                 float* __restrict__ C, int ldc,
                 int M, int N, int K, int act)
{
  const int lane = threadIdx.x & 31;
  const int wv   = threadIdx.x >> 5;
  const int tm   = blockIdx.y * 32 + (wv >> 2) * 16;
  const int tn   = blockIdx.x * 64 + (wv & 3) * 16;
  const int half = lane >> 4;
  const int l15  = lane & 15;
  const int mrow = tm + l15;
  const int ncol = tn + l15;
  const int nld  = (ncol < N) ? ncol : (N - 1);   // clamp loads, guard stores
  const unsigned short* arow = A  + (size_t)mrow * lda;
  const unsigned short* brow = WT + (size_t)nld  * K;

  v8f acc = {0.f,0.f,0.f,0.f,0.f,0.f,0.f,0.f};
  for (int k0 = 0; k0 < K; k0 += 32) {
    HalfVec av, bv;
    // A 16-bit 16x32 layout: lanes hi=0 -> K 0..7 / 16..23 ; hi=1 -> 8..15 / 24..31
    av.u[0] = *reinterpret_cast<const uint4*>(arow + k0 + 8 * half);
    av.u[1] = *reinterpret_cast<const uint4*>(arow + k0 + 16 + 8 * half);
    // B 16-bit 32x16 layout: lanes hi=0 -> K 0..7 / 16..23 ; hi=1 -> 8..15 / 24..31
    bv.u[0] = *reinterpret_cast<const uint4*>(brow + k0 + 8 * half);
    bv.u[1] = *reinterpret_cast<const uint4*>(brow + k0 + 16 + 8 * half);
    if (k0 + 32 < K) {
      __builtin_prefetch(arow + k0 + 32, 0, 0);   // global_prefetch_b8
      __builtin_prefetch(brow + k0 + 32, 0, 0);
    }
    acc = __builtin_amdgcn_wmma_f32_16x16x32_bf16(
        /*neg_a=*/false, av.v, /*neg_b=*/false, bv.v,
        /*c_mod=*/(short)0, acc, /*reuse_a=*/false, /*reuse_b=*/false);
  }
  if (ncol < N) {
    const float bb = bias ? bias[ncol] : 0.0f;
#pragma unroll
    for (int r = 0; r < 8; ++r) {             // C/D layout: m = r + 8*half
      float v = acc[r] + bb;
      if (act == 1) v = tanhf(v);
      else if (act == 2) v = sigmoidf_(v);
      C[(size_t)(tm + r + 8 * half) * ldc + ncol] = v;
    }
  }
}

// ---------------------------------------------------------------------------
// Utility kernels
// ---------------------------------------------------------------------------
__global__ void k_transpose_f2bf(const float* __restrict__ W,   // [K,N] f32
                                 unsigned short* __restrict__ WT, // [N,K] bf16
                                 int K, int N)
{
  size_t total = (size_t)K * N;
  for (size_t i = (size_t)blockIdx.x * blockDim.x + threadIdx.x; i < total;
       i += (size_t)gridDim.x * blockDim.x) {
    size_t k = i / N, n = i % N;
    WT[n * (size_t)K + k] = f2bf(W[i]);
  }
}

__global__ void k_f2bf(const float* __restrict__ src, unsigned short* __restrict__ dst, size_t n) {
  for (size_t i = (size_t)blockIdx.x * blockDim.x + threadIdx.x; i < n;
       i += (size_t)gridDim.x * blockDim.x)
    dst[i] = f2bf(src[i]);
}

__global__ void k_zero_u16(unsigned short* p, size_t n) {
  for (size_t i = (size_t)blockIdx.x * blockDim.x + threadIdx.x; i < n;
       i += (size_t)gridDim.x * blockDim.x) p[i] = 0;
}
__global__ void k_zero_f32(float* p, size_t n) {
  for (size_t i = (size_t)blockIdx.x * blockDim.x + threadIdx.x; i < n;
       i += (size_t)gridDim.x * blockDim.x) p[i] = 0.f;
}

// fuse_feat = v_att * q_att (in place on v_att); also fuse_sum(bf16) and
// x1[:,0:H] = bf16(mean) for t=0.
__global__ __launch_bounds__(256)
void k_fuse(float* __restrict__ fuse, const float* __restrict__ qatt,
            unsigned short* __restrict__ fsum_bf, unsigned short* __restrict__ x1)
{
  int i = blockIdx.x * blockDim.x + threadIdx.x;
  if (i >= KB * KH) return;
  int b = i / KH, h = i % KH;
  float q = qatt[i];
  float s = 0.f;
  for (int r = 0; r < KR; ++r) {
    size_t idx = ((size_t)b * KR + r) * KH + h;
    float f = fuse[idx] * q;
    fuse[idx] = f;
    s += f;
  }
  fsum_bf[i] = f2bf(s);
  x1[(size_t)b * (3 * KH) + h] = f2bf(s / (float)KR);
}

// z = sig(GX[:, :H] + GH[:, :H]); r = sig(GX[:,H:2H] + GH[:,H:2H]); rh = bf16(r*h)
__global__ __launch_bounds__(256)
void k_gru_zr(const float* __restrict__ GX, const float* __restrict__ GH,
              const float* __restrict__ hprev,
              float* __restrict__ z, unsigned short* __restrict__ rh)
{
  int i = blockIdx.x * blockDim.x + threadIdx.x;
  if (i >= KB * KH) return;
  int b = i / KH, h = i % KH;
  float zz = sigmoidf_(GX[(size_t)b * (3 * KH) + h] + GH[(size_t)b * (2 * KH) + h]);
  float rr = sigmoidf_(GX[(size_t)b * (3 * KH) + KH + h] + GH[(size_t)b * (2 * KH) + KH + h]);
  z[i] = zz;
  rh[i] = f2bf(rr * hprev[i]);
}

// m = tanh(GX[:,2H:3H] + HM); hnew = (1-z)*hprev + z*m ; bf16 copy at stride bfld
__global__ __launch_bounds__(256)
void k_gru_out(const float* __restrict__ GX, const float* __restrict__ HM,
               const float* __restrict__ z, const float* __restrict__ hprev,
               float* __restrict__ hnew, unsigned short* __restrict__ bfout, int bfld)
{
  int i = blockIdx.x * blockDim.x + threadIdx.x;
  if (i >= KB * KH) return;
  int b = i / KH, h = i % KH;
  float m  = tanhf(GX[(size_t)b * (3 * KH) + 2 * KH + h] + HM[i]);
  float hv = (1.f - z[i]) * hprev[i] + z[i] * m;
  if (hnew) hnew[i] = hv;
  bfout[(size_t)b * bfld + h] = f2bf(hv);
}

// attention: logits[b,r] = sum_h tanh(hproj[b,h]+fuse[b,r,h])*attw[h] + attb;
// softmax over R; att_x[b,d] = bf16(sum_r p[r]*img[b,r,d])
__global__ __launch_bounds__(256)
void k_att(const float* __restrict__ hproj, const float* __restrict__ fuse,
           const float* __restrict__ attw, const float* __restrict__ attb,
           const float* __restrict__ img, unsigned short* __restrict__ attx)
{
  const int b = blockIdx.x;
  const int lane = threadIdx.x & 31, wv = threadIdx.x >> 5;
  __shared__ float logits[KR];
  const float* hp = hproj + (size_t)b * KH;
  for (int r = wv; r < KR; r += 8) {
    const float* fr = fuse + ((size_t)b * KR + r) * KH;
    float s = 0.f;
    for (int i = lane; i < KH; i += 32) s += tanhf(hp[i] + fr[i]) * attw[i];
    for (int off = 16; off; off >>= 1) s += __shfl_down(s, off);
    if (lane == 0) logits[r] = s + attb[0];
  }
  __syncthreads();
  if (threadIdx.x == 0) {
    float mx = -3.402823466e38f;
    for (int r = 0; r < KR; ++r) mx = fmaxf(mx, logits[r]);
    float sum = 0.f;
    for (int r = 0; r < KR; ++r) { float e = expf(logits[r] - mx); logits[r] = e; sum += e; }
    float inv = 1.0f / sum;
    for (int r = 0; r < KR; ++r) logits[r] *= inv;
  }
  __syncthreads();
  const float* ib = img + (size_t)b * KR * KD;
  for (int d = threadIdx.x; d < KD; d += 256) {
    float s = 0.f;
    for (int r = 0; r < KR; ++r) s += logits[r] * ib[(size_t)r * KD + d];
    attx[(size_t)b * KD + d] = f2bf(s);
  }
}

// x2[:, :H] = bf16(venc*qenc)
__global__ __launch_bounds__(256)
void k_x2(const float* __restrict__ venc, const float* __restrict__ qenc,
          unsigned short* __restrict__ x2)
{
  int i = blockIdx.x * blockDim.x + threadIdx.x;
  if (i >= KB * KH) return;
  int b = i / KH, h = i % KH;
  x2[(size_t)b * (2 * KH) + h] = f2bf(venc[i] * qenc[i]);
}

// gate[b] = sig(h2 . sgw + sgb); sim[b,:] = softmax_r(h2 . vis[b,r,:])
__global__ __launch_bounds__(256)
void k_gate_sim(const float* __restrict__ h2, const float* __restrict__ vis,
                const float* __restrict__ sgw, const float* __restrict__ sgb,
                float* __restrict__ gate, float* __restrict__ sim)
{
  const int b = blockIdx.x;
  const int lane = threadIdx.x & 31, wv = threadIdx.x >> 5;
  const float* h = h2 + (size_t)b * KH;
  __shared__ float wpart[8];
  __shared__ float logits[KR];
  float gs = 0.f;
  for (int i = threadIdx.x; i < KH; i += 256) gs += h[i] * sgw[i];
  for (int off = 16; off; off >>= 1) gs += __shfl_down(gs, off);
  if (lane == 0) wpart[wv] = gs;
  __syncthreads();
  if (threadIdx.x == 0) {
    float t = 0.f;
    for (int w = 0; w < 8; ++w) t += wpart[w];
    gate[b] = sigmoidf_(t + sgb[0]);
  }
  for (int r = wv; r < KR; r += 8) {
    const float* vr = vis + ((size_t)b * KR + r) * KH;
    float s = 0.f;
    for (int i = lane; i < KH; i += 32) s += h[i] * vr[i];
    for (int off = 16; off; off >>= 1) s += __shfl_down(s, off);
    if (lane == 0) logits[r] = s;
  }
  __syncthreads();
  if (threadIdx.x == 0) {
    float mx = -3.402823466e38f;
    for (int r = 0; r < KR; ++r) mx = fmaxf(mx, logits[r]);
    float sum = 0.f;
    for (int r = 0; r < KR; ++r) { float e = expf(logits[r] - mx); logits[r] = e; sum += e; }
    float inv = 1.0f / sum;
    for (int r = 0; r < KR; ++r) sim[(size_t)b * KR + r] = logits[r] * inv;
  }
}

// simpred[b,v] = sum_r sim[b,r]*SM[v,r]
__global__ __launch_bounds__(256)
void k_simpred(const float* __restrict__ sim, const float* __restrict__ sm,
               float* __restrict__ simpred)
{
  int i = blockIdx.x * blockDim.x + threadIdx.x;
  if (i >= KB * KV) return;
  int b = i / KV, v = i % KV;
  const float* smr = sm + (size_t)v * KR;
  const float* sr  = sim + (size_t)b * KR;
  float s = 0.f;
  for (int r = 0; r < KR; ++r) s += sr[r] * smr[r];
  simpred[i] = s;
}

// pred = gate*softmax(L) + (1-gate)*simpred; write out_sent[b,t,:]; argmax -> idx;
// x1[:,2H:3H] = bf16(exp_embed[idx]); x1[:,0:H] = fuse_sum_bf (for next step)
__global__ __launch_bounds__(256)
void k_pred_final(const float* __restrict__ L, const float* __restrict__ simpred,
                  const float* __restrict__ gate, const float* __restrict__ expemb,
                  const unsigned short* __restrict__ fsum_bf,
                  float* __restrict__ out_sent, int t,
                  unsigned short* __restrict__ x1)
{
  const int b = blockIdx.x;
  const int tid = threadIdx.x;
  const float* Lb = L + (size_t)b * KV;
  __shared__ float sv[256];
  __shared__ int   si[256];
  float mx = -3.402823466e38f;
  for (int v = tid; v < KV; v += 256) mx = fmaxf(mx, Lb[v]);
  sv[tid] = mx; __syncthreads();
  for (int s = 128; s > 0; s >>= 1) { if (tid < s) sv[tid] = fmaxf(sv[tid], sv[tid + s]); __syncthreads(); }
  const float rowmax = sv[0]; __syncthreads();
  float sum = 0.f;
  for (int v = tid; v < KV; v += 256) sum += expf(Lb[v] - rowmax);
  sv[tid] = sum; __syncthreads();
  for (int s = 128; s > 0; s >>= 1) { if (tid < s) sv[tid] += sv[tid + s]; __syncthreads(); }
  const float inv = 1.0f / sv[0]; __syncthreads();
  const float g = gate[b];
  const float* sp = simpred + (size_t)b * KV;
  float* ob = out_sent + ((size_t)b * KT + t) * KV;
  float best = -3.402823466e38f; int bi = 0x7fffffff;
  for (int v = tid; v < KV; v += 256) {
    float pw = g * (expf(Lb[v] - rowmax) * inv) + (1.f - g) * sp[v];
    ob[v] = pw;
    if (pw > best) { best = pw; bi = v; }       // strict > : first occurrence in stride
  }
  sv[tid] = best; si[tid] = bi; __syncthreads();
  for (int s = 128; s > 0; s >>= 1) {
    if (tid < s) {
      if (sv[tid + s] > sv[tid] || (sv[tid + s] == sv[tid] && si[tid + s] < si[tid])) {
        sv[tid] = sv[tid + s]; si[tid] = si[tid + s];
      }
    }
    __syncthreads();
  }
  const int idx = si[0];
  for (int h = tid; h < KH; h += 256) {
    x1[(size_t)b * (3 * KH) + 2 * KH + h] = f2bf(expemb[(size_t)idx * KH + h]);
    x1[(size_t)b * (3 * KH) + h]          = fsum_bf[(size_t)b * KH + h];
  }
}

// ---------------------------------------------------------------------------
// Host orchestration
// ---------------------------------------------------------------------------
extern "C" void kernel_launch(void* const* d_in, const int* in_sizes, int n_in,
                              void* d_out, int out_size, void* d_ws, size_t ws_size,
                              hipStream_t stream)
{
  (void)in_sizes; (void)n_in; (void)out_size; (void)ws_size;
  const float* img    = (const float*)d_in[0];
  const float* cls    = (const float*)d_in[1];
  const float* vis    = (const float*)d_in[2];
  const float* smap   = (const float*)d_in[3];
  const float* attvw  = (const float*)d_in[4];   const float* attvb = (const float*)d_in[5];
  const float* attqw  = (const float*)d_in[6];   const float* attqb = (const float*)d_in[7];
  const float* atthw  = (const float*)d_in[8];   const float* atthb = (const float*)d_in[9];
  const float* attw   = (const float*)d_in[10];  const float* attb  = (const float*)d_in[11];
  const float* qfcw   = (const float*)d_in[12];  const float* qfcb  = (const float*)d_in[13];
  const float* vfcw   = (const float*)d_in[14];  const float* vfcb  = (const float*)d_in[15];
  const float* langw  = (const float*)d_in[16];  const float* langb = (const float*)d_in[17];
  const float* sgw    = (const float*)d_in[18];  const float* sgb   = (const float*)d_in[19];
  const float* answ   = (const float*)d_in[20];  const float* ansb  = (const float*)d_in[21];
  const float* expemb = (const float*)d_in[22];
  const float* awx    = (const float*)d_in[23];  const float* awh   = (const float*)d_in[24];
  const float* abx    = (const float*)d_in[25];  const float* abh   = (const float*)d_in[26];
  const float* lwx    = (const float*)d_in[27];  const float* lwh   = (const float*)d_in[28];
  const float* lbx    = (const float*)d_in[29];  const float* lbh   = (const float*)d_in[30];

  // bump allocator over d_ws (256B aligned chunks)
  size_t off = 0;
  auto alloc = [&](size_t bytes) -> void* {
    off = (off + 255) & ~(size_t)255;
    void* p = (char*)d_ws + off;
    off += bytes;
    return p;
  };
  auto aBF = [&](size_t n) { return (unsigned short*)alloc(n * 2); };
  auto aF  = [&](size_t n) { return (float*)alloc(n * 4); };

  // transposed bf16 weights [N,K]
  unsigned short* wt_attv  = aBF((size_t)KH * KD);
  unsigned short* wt_attq  = aBF((size_t)KH * KH);
  unsigned short* wt_atth  = aBF((size_t)KH * KH);
  unsigned short* wt_qfc   = aBF((size_t)KH * KH);
  unsigned short* wt_vfc   = aBF((size_t)KH * KD);
  unsigned short* wt_lang  = aBF((size_t)KV * KH);
  unsigned short* wt_ans   = aBF((size_t)KA * KH);
  unsigned short* wt_awx   = aBF((size_t)(3 * KH) * (3 * KH));
  unsigned short* wt_awh01 = aBF((size_t)(2 * KH) * KH);
  unsigned short* wt_awh2  = aBF((size_t)KH * KH);
  unsigned short* wt_lwx   = aBF((size_t)(3 * KH) * (2 * KH));
  unsigned short* wt_lwh01 = aBF((size_t)(2 * KH) * KH);
  unsigned short* wt_lwh2  = aBF((size_t)KH * KH);
  // activations
  unsigned short* img_bf  = aBF((size_t)KB * KR * KD);
  unsigned short* cls_bf  = aBF((size_t)KB * KH);
  float*          fuse    = aF((size_t)KB * KR * KH);   // v_att then fuse_feat
  float*          qatt    = aF((size_t)KB * KH);
  float*          qenc    = aF((size_t)KB * KH);
  unsigned short* fsum_bf = aBF((size_t)KB * KH);
  unsigned short* x1_bf   = aBF((size_t)KB * 3 * KH);   // [fuse | h2 | prev]
  unsigned short* x2_bf   = aBF((size_t)KB * 2 * KH);   // [fuse_enc | h1]
  float*          GX      = aF((size_t)KB * 3 * KH);
  float*          GH      = aF((size_t)KB * 2 * KH);
  float*          HM      = aF((size_t)KB * KH);
  float*          zb      = aF((size_t)KB * KH);
  unsigned short* rh_bf   = aBF((size_t)KB * KH);
  float*          h2f     = aF((size_t)KB * KH);
  float*          hproj   = aF((size_t)KB * KH);
  unsigned short* attx_bf = aBF((size_t)KB * KD);
  float*          venc    = aF((size_t)KB * KH);
  float*          Lb      = aF((size_t)KB * KV);
  float*          gateb   = aF((size_t)KB);
  float*          simb    = aF((size_t)KB * KR);
  float*          spredb  = aF((size_t)KB * KV);

  auto T2 = [&](const float* W, unsigned short* WT, int K, int N) {
    size_t tot = (size_t)K * N;
    int blocks = (int)((tot + 255) / 256);
    if (blocks > 32768) blocks = 32768;
    k_transpose_f2bf<<<blocks, 256, 0, stream>>>(W, WT, K, N);
  };
  auto GEMM = [&](const unsigned short* A, int lda, const unsigned short* WT,
                  const float* bias, float* C, int ldc, int M, int N, int K, int act) {
    dim3 g((unsigned)((N + 63) / 64), (unsigned)(M / 32));
    k_gemm_bf16<<<g, 256, 0, stream>>>(A, lda, WT, bias, C, ldc, M, N, K, act);
  };

  // ---- one-time setup ----
  T2(attvw, wt_attv, KD, KH);
  T2(attqw, wt_attq, KH, KH);
  T2(atthw, wt_atth, KH, KH);
  T2(qfcw,  wt_qfc,  KH, KH);
  T2(vfcw,  wt_vfc,  KD, KH);
  T2(langw, wt_lang, KH, KV);
  T2(answ,  wt_ans,  KH, KA);
  for (int g = 0; g < 3; ++g)
    T2(awx + (size_t)g * (3 * KH) * KH, wt_awx + (size_t)g * KH * (3 * KH), 3 * KH, KH);
  for (int g = 0; g < 2; ++g)
    T2(awh + (size_t)g * KH * KH, wt_awh01 + (size_t)g * KH * KH, KH, KH);
  T2(awh + (size_t)2 * KH * KH, wt_awh2, KH, KH);
  for (int g = 0; g < 3; ++g)
    T2(lwx + (size_t)g * (2 * KH) * KH, wt_lwx + (size_t)g * KH * (2 * KH), 2 * KH, KH);
  for (int g = 0; g < 2; ++g)
    T2(lwh + (size_t)g * KH * KH, wt_lwh01 + (size_t)g * KH * KH, KH, KH);
  T2(lwh + (size_t)2 * KH * KH, wt_lwh2, KH, KH);

  k_f2bf<<<32768, 256, 0, stream>>>(img, img_bf, (size_t)KB * KR * KD);
  k_f2bf<<<768, 256, 0, stream>>>(cls, cls_bf, (size_t)KB * KH);

  // v_att = tanh(img @ att_v_w + b)   -> fuse buffer
  GEMM(img_bf, KD, wt_attv, attvb, fuse, KH, KB * KR, KH, KD, 1);
  GEMM(cls_bf, KH, wt_attq, attqb, qatt, KH, KB, KH, KH, 1);   // q_att
  GEMM(cls_bf, KH, wt_qfc,  qfcb,  qenc, KH, KB, KH, KH, 1);   // q_enc

  k_zero_u16<<<2304, 256, 0, stream>>>(x1_bf, (size_t)KB * 3 * KH);
  k_zero_f32<<<768, 256, 0, stream>>>(h2f, (size_t)KB * KH);
  k_fuse<<<768, 256, 0, stream>>>(fuse, qatt, fsum_bf, x1_bf);

  // output_ans = cls @ ans_w + ans_b  (first KB*KA floats of d_out)
  GEMM(cls_bf, KH, wt_ans, ansb, (float*)d_out, KA, KB, KA, KH, 0);
  float* out_sent = (float*)d_out + (size_t)KB * KA;

  // ---- T=12 decode steps ----
  for (int t = 0; t < KT; ++t) {
    // attention GRU (h1 = gru(x1, h2))
    GEMM(x1_bf, 3 * KH, wt_awx, abx, GX, 3 * KH, KB, 3 * KH, 3 * KH, 0);
    GEMM(x1_bf + KH, 3 * KH, wt_awh01, abh, GH, 2 * KH, KB, 2 * KH, KH, 0);
    k_gru_zr<<<768, 256, 0, stream>>>(GX, GH, h2f, zb, rh_bf);
    GEMM(rh_bf, KH, wt_awh2, abh + 2 * KH, HM, KH, KB, KH, KH, 0);
    k_gru_out<<<768, 256, 0, stream>>>(GX, HM, zb, h2f, nullptr, x2_bf + KH, 2 * KH); // h1 -> x2[:,H:]
    // attention over regions
    GEMM(x2_bf + KH, 2 * KH, wt_atth, atthb, hproj, KH, KB, KH, KH, 0);
    k_att<<<KB, 256, 0, stream>>>(hproj, fuse, attw, attb, img, attx_bf);
    GEMM(attx_bf, KD, wt_vfc, vfcb, venc, KH, KB, KH, KD, 1);   // v_enc = tanh(...)
    k_x2<<<768, 256, 0, stream>>>(venc, qenc, x2_bf);           // x2[:, :H] = v_enc*q_enc
    // language GRU (h2 = gru(x2, h2))
    GEMM(x2_bf, 2 * KH, wt_lwx, lbx, GX, 3 * KH, KB, 3 * KH, 2 * KH, 0);
    GEMM(x1_bf + KH, 3 * KH, wt_lwh01, lbh, GH, 2 * KH, KB, 2 * KH, KH, 0);
    k_gru_zr<<<768, 256, 0, stream>>>(GX, GH, h2f, zb, rh_bf);
    GEMM(rh_bf, KH, wt_lwh2, lbh + 2 * KH, HM, KH, KB, KH, KH, 0);
    k_gru_out<<<768, 256, 0, stream>>>(GX, HM, zb, h2f, h2f, x1_bf + KH, 3 * KH); // h2 -> x1[:,H:2H]
    // word distribution
    GEMM(x1_bf + KH, 3 * KH, wt_lang, langb, Lb, KV, KB, KV, KH, 0);
    k_gate_sim<<<KB, 256, 0, stream>>>(h2f, vis, sgw, sgb, gateb, simb);
    k_simpred<<<(KB * KV + 255) / 256, 256, 0, stream>>>(simb, smap, spredb);
    k_pred_final<<<KB, 256, 0, stream>>>(Lb, spredb, gateb, expemb, fsum_bf,
                                         out_sent, t, x1_bf);
  }
}
